// VideoEncoderLSTM_82841329205708
// MI455X (gfx1250) — compile-verified, matrix operand order checked by reading, not run on hardware
//
#include <hip/hip_runtime.h>
#include <cstdint>

#ifndef __has_builtin
#define __has_builtin(x) 0
#endif

// ---------------------------------------------------------------------------
// WMMA bf16 helpers (gfx1250, wave32)
// ---------------------------------------------------------------------------
typedef __bf16   v16bf __attribute__((ext_vector_type(16)));
typedef float    v8f   __attribute__((ext_vector_type(8)));
typedef unsigned v4u   __attribute__((ext_vector_type(4)));
typedef unsigned u4    __attribute__((ext_vector_type(4)));
typedef int      v8i   __attribute__((ext_vector_type(8)));
typedef int      v4i   __attribute__((ext_vector_type(4)));

union Frag { unsigned u[8]; u4 q[2]; v16bf v; };

__device__ inline v8f wmma_bf16(const Frag& a, const Frag& b, v8f c) {
    return __builtin_amdgcn_wmma_f32_16x16x32_bf16(false, a.v, false, b.v,
                                                   (short)0, c, false, false);
}

// Per the ISA 16-bit A/B layout, a lane's 16 bf16 fragment elements are two
// contiguous 8-element (16B) runs at k = hi*8 and k = 16 + hi*8 within the
// 32-wide K chunk -> exactly two b128 loads.
__device__ inline void ldfrag(Frag& f, const unsigned short* chunk, int hi) {
    f.q[0] = *(const u4*)(chunk + hi * 8);
    f.q[1] = *(const u4*)(chunk + 16 + hi * 8);
}

__device__ inline unsigned short f2bf(float f) {
    unsigned u = __builtin_bit_cast(unsigned, f);
    unsigned r = u + 0x7FFFu + ((u >> 16) & 1u);
    return (unsigned short)(r >> 16);
}
__device__ inline float sigf(float x) { return 1.0f / (1.0f + expf(-x)); }

// ---------------------------------------------------------------------------
// Conversion kernels (f32 -> bf16, with layout permutations)
// ---------------------------------------------------------------------------
__global__ void k_cvt(const float* __restrict__ s, unsigned short* __restrict__ d, int n) {
    for (int i = blockIdx.x * blockDim.x + threadIdx.x; i < n; i += gridDim.x * blockDim.x)
        d[i] = f2bf(s[i]);
}

// conv weights [OC=64][IC][3][3] -> [oc][tap][ic] bf16
__global__ void k_cvt_convw(const float* __restrict__ s, unsigned short* __restrict__ d, int IC) {
    int n = 64 * IC * 9;
    for (int i = blockIdx.x * blockDim.x + threadIdx.x; i < n; i += gridDim.x * blockDim.x) {
        int oc = i / (IC * 9);
        int r  = i - oc * IC * 9;
        int tap = r / IC;
        int ic  = r - tap * IC;
        d[i] = f2bf(s[(oc * IC + ic) * 9 + tap]);
    }
}

// fc1_w [1024][65536] with column perm (c*1024+pix) -> (pix*64+c), bf16 [n][k']
__global__ void k_cvt_fc1(const float* __restrict__ s, unsigned short* __restrict__ d) {
    long long n = 1024LL * 65536LL;
    for (long long i = (long long)blockIdx.x * blockDim.x + threadIdx.x; i < n;
         i += (long long)gridDim.x * blockDim.x) {
        int nr = (int)(i >> 16), k = (int)(i & 65535);
        int pix = k >> 6, c = k & 63;
        d[i] = f2bf(s[(long long)nr * 65536 + c * 1024 + pix]);
    }
}

// f_t [B][T][C][32][32] -> A rows 256..511 as [img][pix*64+c] bf16
__global__ void k_app(const float* __restrict__ f_t, unsigned short* __restrict__ A) {
    long long n = 256LL * 65536LL;
    for (long long i = (long long)blockIdx.x * blockDim.x + threadIdx.x; i < n;
         i += (long long)gridDim.x * blockDim.x) {
        int img = (int)(i >> 16), k = (int)(i & 65535);
        int pix = k >> 6, c = k & 63;
        A[(long long)(256 + img) * 65536 + k] =
            f2bf(f_t[(long long)(img * 64 + c) * 1024 + pix]);
    }
}

// ---------------------------------------------------------------------------
// Conv1: 3x3, 128->64, pad1, ReLU. One block per image, implicit GEMM (WMMA).
// LDS: fully padded [34][34][128] bf16 (halo zeroed) -> no predicates at all.
// out m1[img][pix][oc] bf16
// ---------------------------------------------------------------------------
__global__ __launch_bounds__(256) void k_conv1(const float* __restrict__ f_t,
                                               const float* __restrict__ f_t_1,
                                               const unsigned short* __restrict__ wb,
                                               const float* __restrict__ bias,
                                               unsigned short* __restrict__ out) {
    extern __shared__ char dsm[];
    unsigned short* lds = (unsigned short*)dsm;   // [34][34][128]
    const int img = blockIdx.x;
    // zero everything (halo included), then fill interior
    unsigned* z = (unsigned*)lds;
    for (int i = threadIdx.x; i < 34 * 34 * 64; i += 256) z[i] = 0u;
    __syncthreads();
    const float4* s0 = (const float4*)(f_t   + (long long)img * 64 * 1024);
    const float4* s1 = (const float4*)(f_t_1 + (long long)img * 64 * 1024);
    for (int i = threadIdx.x; i < 64 * 256; i += 256) {  // 4 pixels / iter
        int c = i >> 8, p4 = (i & 255) * 4;
        int y = p4 >> 5, x = p4 & 31;
        float4 a = s0[i];
        float4 b = s1[i];
        int o = ((y + 1) * 34 + (x + 1)) * 128;
        lds[o + c]             = f2bf(a.x); lds[o + 64 + c]       = f2bf(b.x);
        lds[o + 128 + c]       = f2bf(a.y); lds[o + 192 + c]      = f2bf(b.y);
        lds[o + 256 + c]       = f2bf(a.z); lds[o + 320 + c]      = f2bf(b.z);
        lds[o + 384 + c]       = f2bf(a.w); lds[o + 448 + c]      = f2bf(b.w);
    }
    __syncthreads();
    const int lane = threadIdx.x & 31, w = threadIdx.x >> 5;
    const int hi = lane >> 4, l15 = lane & 15;
    for (int tile = w; tile < 256; tile += 8) {
        int mt = tile >> 2, nt = tile & 3;
        int y = mt >> 1;
        int x = ((mt * 16) & 31) + l15;
        v8f acc; for (int r = 0; r < 8; ++r) acc[r] = 0.f;
        for (int kc = 0; kc < 36; ++kc) {
            int tap = kc >> 2, ic0 = (kc & 3) * 32;
            int ky = tap / 3, kx = tap - ky * 3;
            const unsigned short* arow = &lds[((y + ky) * 34 + (x + kx)) * 128 + ic0];
            const unsigned short* brow = &wb[(nt * 16 + l15) * 1152 + tap * 128 + ic0];
            Frag a, b;
            ldfrag(a, arow, hi);
            ldfrag(b, brow, hi);
            acc = wmma_bf16(a, b, acc);
        }
#pragma unroll
        for (int r = 0; r < 8; ++r) {
            int m = r + hi * 8;
            int pix = mt * 16 + m;
            int oc = nt * 16 + l15;
            float v = acc[r] + bias[oc];
            v = v > 0.f ? v : 0.f;
            out[((long long)img * 1024 + pix) * 64 + oc] = f2bf(v);
        }
    }
}

// ---------------------------------------------------------------------------
// Conv2: 3x3, 64->64, pad1, ReLU. Input staged to LDS with the Tensor Data
// Mover (one contiguous 128KB tile per image); x-border handled by a pointer
// select to a zeroed LDS scratch line (v_cndmask, no exec masking).
// out -> Abuf rows [0..255] = mot activations in fc1 layout.
// ---------------------------------------------------------------------------
__global__ __launch_bounds__(256) void k_conv2(const unsigned short* __restrict__ m1,
                                               const unsigned short* __restrict__ wb,
                                               const float* __restrict__ bias,
                                               unsigned short* __restrict__ A) {
    extern __shared__ char dsm[];
    unsigned short* lds = (unsigned short*)dsm;        // [34][32][64] + 64 zero scratch
    unsigned short* zbuf = lds + 34 * 32 * 64;
    const int img = blockIdx.x;
    for (int i = threadIdx.x; i < 32 * 64; i += 256) { lds[i] = 0; lds[33 * 32 * 64 + i] = 0; }
    for (int i = threadIdx.x; i < 64; i += 256) zbuf[i] = 0;
    __syncthreads();
    const unsigned short* src = m1 + (long long)img * 65536;
#if __has_builtin(__builtin_amdgcn_tensor_load_to_lds) && __has_builtin(__builtin_amdgcn_s_wait_tensorcnt)
    if (threadIdx.x < 32) {   // one wave issues the TDM op (EXEC-independent)
        unsigned long long ga = (unsigned long long)(size_t)src;
        v4u g0;
        g0[0] = 1u;                       // count=1, user descriptor
        g0[1] = 4096u;                    // lds_addr = row 1 (past y-halo)
        g0[2] = (unsigned)ga;             // global_addr[31:0]
        g0[3] = ((unsigned)(ga >> 32) & 0x01FFFFFFu) | 0x80000000u;  // addr[56:32] | type=2
        v8i g1;
        for (int q = 0; q < 8; ++q) g1[q] = 0;
        g1[0] = 0x30000;                  // data_size = 8B, no multicast
        g1[1] = (int)(16384u << 16);      // tensor_dim0 = 16384 (lo16 in [31:16])
        g1[2] = (1 << 16);                // tensor_dim0 hi16=0, tensor_dim1 = 1
        g1[3] = (int)(16384u << 16);      // tile_dim0 = 16384 elements (128KB)
        g1[5] = 16384;                    // tensor_dim0_stride
        v4i g2; for (int q = 0; q < 4; ++q) g2[q] = 0;
        v4i g3; for (int q = 0; q < 4; ++q) g3[q] = 0;
#if defined(__clang_major__) && (__clang_major__ >= 23)
        v8i g4; for (int q = 0; q < 8; ++q) g4[q] = 0;
        __builtin_amdgcn_tensor_load_to_lds(g0, g1, g2, g3, g4, 0);
#else
        __builtin_amdgcn_tensor_load_to_lds(g0, g1, g2, g3, 0);
#endif
        __builtin_amdgcn_s_wait_tensorcnt(0);
    }
#else
    {
        unsigned* dst32 = (unsigned*)(lds + 2048);
        const unsigned* s32 = (const unsigned*)src;
        for (int i = threadIdx.x; i < 32768; i += 256) dst32[i] = s32[i];
    }
#endif
    __syncthreads();
    const int lane = threadIdx.x & 31, w = threadIdx.x >> 5;
    const int hi = lane >> 4, l15 = lane & 15;
    for (int tile = w; tile < 256; tile += 8) {
        int mt = tile >> 2, nt = tile & 3;
        int y = mt >> 1;
        int x = ((mt * 16) & 31) + l15;
        v8f acc; for (int r = 0; r < 8; ++r) acc[r] = 0.f;
        for (int kc = 0; kc < 18; ++kc) {
            int tap = kc >> 1, ic0 = (kc & 1) * 32;
            int ky = tap / 3, kx = tap - ky * 3;
            int xs = x + kx - 1;
            const unsigned short* arow = ((unsigned)xs < 32u)
                ? &lds[((y + ky) * 32 + xs) * 64 + ic0] : zbuf;
            const unsigned short* brow = &wb[(nt * 16 + l15) * 576 + tap * 64 + ic0];
            Frag a, b;
            ldfrag(a, arow, hi);
            ldfrag(b, brow, hi);
            acc = wmma_bf16(a, b, acc);
        }
#pragma unroll
        for (int r = 0; r < 8; ++r) {
            int m = r + hi * 8;
            int pix = mt * 16 + m;
            int oc = nt * 16 + l15;
            float v = acc[r] + bias[oc];
            v = v > 0.f ? v : 0.f;
            A[(long long)img * 65536 + pix * 64 + oc] = f2bf(v);
        }
    }
}

// ---------------------------------------------------------------------------
// Generic LDS-tiled bf16 GEMM: out = act(A[M][K] * B[N][K]^T + bias[N])
// 64x64 block, 8 waves (wm 0..3 x wn 0..1, 2 N-tiles each).
// ---------------------------------------------------------------------------
template <bool RELU, bool OUTBF>
__global__ __launch_bounds__(256) void k_gemm(const unsigned short* __restrict__ A,
                                              const unsigned short* __restrict__ B,
                                              const float* __restrict__ bias,
                                              void* __restrict__ out,
                                              int M, int N, int K) {
    __shared__ unsigned short lA[64 * 64];
    __shared__ unsigned short lB[64 * 64];
    const int m0 = blockIdx.y * 64, n0 = blockIdx.x * 64;
    const int lane = threadIdx.x & 31, w = threadIdx.x >> 5;
    const int hi = lane >> 4, l15 = lane & 15;
    const int wm = w & 3, wn = w >> 2;
    v8f acc0, acc1;
    for (int r = 0; r < 8; ++r) { acc0[r] = 0.f; acc1[r] = 0.f; }
    u4* ua = (u4*)lA;
    u4* ub = (u4*)lB;
    for (int k0 = 0; k0 < K; k0 += 64) {
        __syncthreads();
        // 64 rows x 64 bf16 cols = 512 x 16B per panel; 2 x u4 per thread
        for (int i = threadIdx.x; i < 512; i += 256) {
            int row = i >> 3, col = (i & 7) * 8;
            ua[i] = *(const u4*)(A + (long long)(m0 + row) * K + k0 + col);
            ub[i] = *(const u4*)(B + (long long)(n0 + row) * K + k0 + col);
        }
        __syncthreads();
#pragma unroll
        for (int ks = 0; ks < 2; ++ks) {
            Frag a;
            ldfrag(a, lA + (wm * 16 + l15) * 64 + ks * 32, hi);
#pragma unroll
            for (int j = 0; j < 2; ++j) {
                Frag b;
                int nr = (wn * 2 + j) * 16 + l15;
                ldfrag(b, lB + nr * 64 + ks * 32, hi);
                if (j == 0) acc0 = wmma_bf16(a, b, acc0);
                else        acc1 = wmma_bf16(a, b, acc1);
            }
        }
    }
#pragma unroll
    for (int j = 0; j < 2; ++j) {
        v8f acc = j ? acc1 : acc0;
        int n = n0 + (wn * 2 + j) * 16 + l15;
#pragma unroll
        for (int r = 0; r < 8; ++r) {
            int m = m0 + wm * 16 + r + hi * 8;
            float v = acc[r] + bias[n];
            if (RELU) v = v > 0.f ? v : 0.f;
            if (OUTBF) ((unsigned short*)out)[(long long)m * N + n] = f2bf(v);
            else       ((float*)out)[(long long)m * N + n] = v;
        }
    }
}

// ---------------------------------------------------------------------------
// Attention / gating: alpha, and time-major bf16 LSTM inputs.
// ---------------------------------------------------------------------------
__global__ void k_att(const float* __restrict__ fc2o, const float* __restrict__ att1,
                      const float* __restrict__ att2, float* __restrict__ alpha,
                      unsigned short* __restrict__ inm, unsigned short* __restrict__ ina) {
    for (int i = blockIdx.x * blockDim.x + threadIdx.x; i < 256 * 512;
         i += gridDim.x * blockDim.x) {
        int img = i >> 9, j = i & 511;
        int b = img >> 4, t = img & 15;
        float mot = fc2o[img * 512 + j];
        float app = fc2o[(256 + img) * 512 + j];
        float al = sigf(att1[b * 512 + j] * mot);
        float aa = sigf(att2[b * 512 + j] * app);
        alpha[img * 512 + j] = al;        // [B][T][512], img = b*16+t
        int o = t * 16384 + b * 1024 + j; // time-major [T][B][1024]
        inm[o] = f2bf(al * mot); inm[o + 512] = f2bf(mot);
        ina[o] = f2bf(aa * app); ina[o + 512] = f2bf(app);
    }
}

// ---------------------------------------------------------------------------
// LSTM: single persistent workgroup, WMMA gate GEMMs (M=16, N=2048).
// ---------------------------------------------------------------------------
__device__ inline void gate_gemm(float* __restrict__ gates,
                                 const unsigned short* __restrict__ x, int Kx,
                                 const unsigned short* __restrict__ wih,
                                 const unsigned short* __restrict__ h,
                                 const unsigned short* __restrict__ whh,
                                 const float* __restrict__ bih,
                                 const float* __restrict__ bhh,
                                 int lane, int w) {
    const int hi = lane >> 4, l15 = lane & 15;
    for (int nt = w; nt < 128; nt += 8) {
        v8f acc; for (int r = 0; r < 8; ++r) acc[r] = 0.f;
        const unsigned short* ax = x + l15 * Kx;
        const unsigned short* bx = wih + (long long)(nt * 16 + l15) * Kx;
        for (int kc = 0; kc < (Kx >> 5); ++kc) {
            Frag a, b;
            ldfrag(a, ax + kc * 32, hi);
            ldfrag(b, bx + kc * 32, hi);
            acc = wmma_bf16(a, b, acc);
        }
        const unsigned short* ah = h + l15 * 512;
        const unsigned short* bh = whh + (nt * 16 + l15) * 512;
        for (int kc = 0; kc < 16; ++kc) {
            Frag a, b;
            ldfrag(a, ah + kc * 32, hi);
            ldfrag(b, bh + kc * 32, hi);
            acc = wmma_bf16(a, b, acc);
        }
#pragma unroll
        for (int r = 0; r < 8; ++r) {
            int m = r + hi * 8;
            int n = nt * 16 + l15;
            gates[m * 2048 + n] = acc[r] + bih[n] + bhh[n];
        }
    }
}

__device__ inline void lstm_update(const float* __restrict__ gates, float* __restrict__ c,
                                   unsigned short* __restrict__ snew,
                                   float* __restrict__ outb, int t, int off) {
    for (int i = threadIdx.x; i < 16 * 512; i += 256) {
        int m = i >> 9, j = i & 511;
        float gi = gates[m * 2048 + j];
        float gf = gates[m * 2048 + 512 + j];
        float gg = gates[m * 2048 + 1024 + j];
        float go = gates[m * 2048 + 1536 + j];
        float c2 = sigf(gf) * c[i] + sigf(gi) * tanhf(gg);
        float hh = sigf(go) * tanhf(c2);
        c[i] = c2;
        if (snew) snew[i] = f2bf(hh);
        outb[(long long)(m * 16 + t) * 1024 + off + j] = hh;
    }
}

__global__ __launch_bounds__(256) void k_lstm(
    const unsigned short* __restrict__ inm, const unsigned short* __restrict__ ina,
    const unsigned short* w1m_ih, const unsigned short* w1m_hh, const float* b1m_ih, const float* b1m_hh,
    const unsigned short* w2m_ih, const unsigned short* w2m_hh, const float* b2m_ih, const float* b2m_hh,
    const unsigned short* w1a_ih, const unsigned short* w1a_hh, const float* b1a_ih, const float* b1a_hh,
    const unsigned short* w2a_ih, const unsigned short* w2a_hh, const float* b2a_ih, const float* b2a_hh,
    float* __restrict__ out_a, float* __restrict__ out_m) {
    extern __shared__ char dsm[];
    float* gates = (float*)dsm;                       // 16*2048 f32 = 128KB
    float* c1m = (float*)(dsm + 131072);
    float* c2m = c1m + 16 * 512;
    float* c1a = c2m + 16 * 512;
    float* c2a = c1a + 16 * 512;
    unsigned short* s1m = (unsigned short*)(c2a + 16 * 512);
    unsigned short* s1a = s1m + 16 * 512;
    unsigned short* s2a = s1a + 16 * 512;
    for (int i = threadIdx.x; i < 16 * 512; i += 256) {
        c1m[i] = 0.f; c2m[i] = 0.f; c1a[i] = 0.f; c2a[i] = 0.f;
        s1m[i] = 0; s1a[i] = 0; s2a[i] = 0;
    }
    const int lane = threadIdx.x & 31, w = threadIdx.x >> 5;
    __syncthreads();
    for (int t = 0; t < 16; ++t) {
        // l1m: x = in_m[t] (K=1024), h = s1m
        gate_gemm(gates, inm + t * 16384, 1024, w1m_ih, s1m, w1m_hh, b1m_ih, b1m_hh, lane, w);
        __syncthreads();
        lstm_update(gates, c1m, s1m, out_m, t, 0);
        __syncthreads();
        // l2m: x = h = new s1m (K=512), c = c2m; s2m is output-only
        gate_gemm(gates, s1m, 512, w2m_ih, s1m, w2m_hh, b2m_ih, b2m_hh, lane, w);
        __syncthreads();
        lstm_update(gates, c2m, nullptr, out_m, t, 512);
        __syncthreads();
        // l1a: x = in_a[t] (K=1024), h = s1a
        gate_gemm(gates, ina + t * 16384, 1024, w1a_ih, s1a, w1a_hh, b1a_ih, b1a_hh, lane, w);
        __syncthreads();
        lstm_update(gates, c1a, s1a, out_a, t, 0);
        __syncthreads();
        // l2a: x = new s1a, h = old s2a, c = c2a -> new s2a
        gate_gemm(gates, s1a, 512, w2a_ih, s2a, w2a_hh, b2a_ih, b2a_hh, lane, w);
        __syncthreads();
        lstm_update(gates, c2a, s2a, out_a, t, 512);
        __syncthreads();
    }
}

// ---------------------------------------------------------------------------
// Orchestration
// ---------------------------------------------------------------------------
extern "C" void kernel_launch(void* const* d_in, const int* in_sizes, int n_in,
                              void* d_out, int out_size, void* d_ws, size_t ws_size,
                              hipStream_t stream) {
    (void)in_sizes; (void)n_in; (void)out_size; (void)ws_size;
    const float* f_t    = (const float*)d_in[0];
    const float* f_t_1  = (const float*)d_in[1];
    const float* att1   = (const float*)d_in[2];
    const float* att2   = (const float*)d_in[3];
    const float* conv1w = (const float*)d_in[4];
    const float* conv1b = (const float*)d_in[5];
    const float* conv2w = (const float*)d_in[6];
    const float* conv2b = (const float*)d_in[7];
    const float* l1a_wih = (const float*)d_in[8];  const float* l1a_whh = (const float*)d_in[9];
    const float* l1a_bih = (const float*)d_in[10]; const float* l1a_bhh = (const float*)d_in[11];
    const float* l2a_wih = (const float*)d_in[12]; const float* l2a_whh = (const float*)d_in[13];
    const float* l2a_bih = (const float*)d_in[14]; const float* l2a_bhh = (const float*)d_in[15];
    const float* l1m_wih = (const float*)d_in[16]; const float* l1m_whh = (const float*)d_in[17];
    const float* l1m_bih = (const float*)d_in[18]; const float* l1m_bhh = (const float*)d_in[19];
    const float* l2m_wih = (const float*)d_in[20]; const float* l2m_whh = (const float*)d_in[21];
    const float* l2m_bih = (const float*)d_in[22]; const float* l2m_bhh = (const float*)d_in[23];
    const float* fc1w = (const float*)d_in[24];
    const float* fc1b = (const float*)d_in[25];
    const float* fc2w = (const float*)d_in[26];
    const float* fc2b = (const float*)d_in[27];

    float* out  = (float*)d_out;
    float* outa = out;                 // [B][T][1024]
    float* outm = out + 262144;        // [B][T][1024]
    float* alph = out + 524288;        // [B][T][512]

    size_t off = 0;
    auto alloc = [&](size_t b) -> void* {
        void* p = (char*)d_ws + off;
        off += (b + 255) & ~(size_t)255;
        return p;
    };
    unsigned short* Abuf   = (unsigned short*)alloc(512ull * 65536 * 2);   // mot|app rows
    unsigned short* m1     = (unsigned short*)alloc(256ull * 1024 * 64 * 2);
    unsigned short* fc1wb  = (unsigned short*)alloc(1024ull * 65536 * 2);
    unsigned short* fc2wb  = (unsigned short*)alloc(512ull * 1024 * 2);
    unsigned short* c1wb   = (unsigned short*)alloc(64ull * 1152 * 2);
    unsigned short* c2wb   = (unsigned short*)alloc(64ull * 576 * 2);
    unsigned short* b1m_ih = (unsigned short*)alloc(2048ull * 1024 * 2);
    unsigned short* b1m_hh = (unsigned short*)alloc(2048ull * 512 * 2);
    unsigned short* b2m_ih = (unsigned short*)alloc(2048ull * 512 * 2);
    unsigned short* b2m_hh = (unsigned short*)alloc(2048ull * 512 * 2);
    unsigned short* b1a_ih = (unsigned short*)alloc(2048ull * 1024 * 2);
    unsigned short* b1a_hh = (unsigned short*)alloc(2048ull * 512 * 2);
    unsigned short* b2a_ih = (unsigned short*)alloc(2048ull * 512 * 2);
    unsigned short* b2a_hh = (unsigned short*)alloc(2048ull * 512 * 2);
    unsigned short* fc1out = (unsigned short*)alloc(512ull * 1024 * 2);
    float*          fc2out = (float*)alloc(512ull * 512 * 4);
    unsigned short* inm    = (unsigned short*)alloc(16ull * 16 * 1024 * 2);
    unsigned short* ina    = (unsigned short*)alloc(16ull * 16 * 1024 * 2);

    // Weight conversions
    k_cvt_fc1<<<8192, 256, 0, stream>>>(fc1w, fc1wb);
    k_cvt<<<2048, 256, 0, stream>>>(fc2w, fc2wb, 512 * 1024);
    k_cvt_convw<<<64, 256, 0, stream>>>(conv1w, c1wb, 128);
    k_cvt_convw<<<32, 256, 0, stream>>>(conv2w, c2wb, 64);
    k_cvt<<<2048, 256, 0, stream>>>(l1m_wih, b1m_ih, 2048 * 1024);
    k_cvt<<<2048, 256, 0, stream>>>(l1m_whh, b1m_hh, 2048 * 512);
    k_cvt<<<2048, 256, 0, stream>>>(l2m_wih, b2m_ih, 2048 * 512);
    k_cvt<<<2048, 256, 0, stream>>>(l2m_whh, b2m_hh, 2048 * 512);
    k_cvt<<<2048, 256, 0, stream>>>(l1a_wih, b1a_ih, 2048 * 1024);
    k_cvt<<<2048, 256, 0, stream>>>(l1a_whh, b1a_hh, 2048 * 512);
    k_cvt<<<2048, 256, 0, stream>>>(l2a_wih, b2a_ih, 2048 * 512);
    k_cvt<<<2048, 256, 0, stream>>>(l2a_whh, b2a_hh, 2048 * 512);
    k_app<<<8192, 256, 0, stream>>>(f_t, Abuf);

    // Convs (implicit GEMM, WMMA)
    k_conv1<<<256, 256, 34 * 34 * 128 * 2, stream>>>(f_t, f_t_1, c1wb, conv1b, m1);
    k_conv2<<<256, 256, (34 * 32 * 64 + 64) * 2, stream>>>(m1, c2wb, conv2b, Abuf);

    // fc1: [512 x 65536] * [1024 x 65536]^T, ReLU, bf16 out
    dim3 g1(1024 / 64, 512 / 64);
    k_gemm<true, true><<<g1, 256, 0, stream>>>(Abuf, fc1wb, fc1b, fc1out, 512, 1024, 65536);
    // fc2: [512 x 1024] * [512 x 1024]^T, f32 out
    dim3 g2(512 / 64, 512 / 64);
    k_gemm<false, false><<<g2, 256, 0, stream>>>(fc1out, fc2wb, fc2b, fc2out, 512, 512, 1024);

    // attention/gating + LSTM inputs
    k_att<<<512, 256, 0, stream>>>(fc2out, att1, att2, alph, inm, ina);

    // recurrent LSTM stack (single persistent WG, 311296B LDS)
    k_lstm<<<1, 256, 311296, stream>>>(inm, ina,
        b1m_ih, b1m_hh, l1m_bih, l1m_bhh,
        b2m_ih, b2m_hh, l2m_bih, l2m_bhh,
        b1a_ih, b1a_hh, l1a_bih, l1a_bhh,
        b2a_ih, b2a_hh, l2a_bih, l2a_bhh,
        outa, outm);
}
